// S_H_Attention2_67783173865476
// MI455X (gfx1250) — compile-verified, matrix operand order checked by reading, not run on hardware
//
#include <hip/hip_runtime.h>

// ---------------------------------------------------------------------------
// MI455X (gfx1250, wave32) implementation.
// Compute-bound workload (~24 GFLOP vs ~38MB input): conv1/conv2/FC/attn1 are
// mapped to v_wmma_f32_16x16x32_f16 implicit GEMMs with f32 accumulation.
// Conv im2col gathers are branch-free (K-padding lands in zeroed LDS regions,
// per-k LDS byte offsets from a precomputed u16 table), and C-tile epilogues
// are branch-free via a clamped scratch row (no saveexec chains).
// ---------------------------------------------------------------------------

typedef _Float16 h16;
typedef __attribute__((ext_vector_type(16))) _Float16       v16h;
typedef __attribute__((ext_vector_type(8)))  _Float16       v8h;
typedef __attribute__((ext_vector_type(4)))  _Float16       v4h;
typedef __attribute__((ext_vector_type(8)))  float          v8f;
typedef __attribute__((ext_vector_type(8)))  unsigned short u16x8;

__device__ __forceinline__ v8f wmma16(v16h a, v16h b, v8f c) {
  // (neg_a, A, neg_b, B, c_mod, C, reuse_a, reuse_b)
  return __builtin_amdgcn_wmma_f32_16x16x32_f16(false, a, false, b, (short)0, c,
                                                false, false);
}

// A/B fragment from a K-contiguous f16 matrix in LDS.
// Lane l: row = r0 + (l&15); K chunks {kb..kb+7, kb+16..kb+23}, kb=k0+8*(l>>4).
__device__ __forceinline__ v16h ldfrag(const h16* base, int ld, int r0, int k0,
                                       int lane) {
  int r  = r0 + (lane & 15);
  int kb = k0 + ((lane >> 4) << 3);
  const h16* q = base + r * ld + kb;
  v8h lo = *(const v8h*)(q);
  v8h hi = *(const v8h*)(q + 16);
  return __builtin_shufflevector(lo, hi, 0,1,2,3,4,5,6,7,8,9,10,11,12,13,14,15);
}

// Branch-free im2col gather: 16 elements at byte offsets pb2+o[i] from base.
__device__ __forceinline__ v16h gather16(const char* base, int pb2, u16x8 o0,
                                         u16x8 o1) {
  v16h f;
#pragma unroll
  for (int i = 0; i < 8; ++i) {
    f[i]     = *(const h16*)(base + pb2 + (int)o0[i]);
    f[i + 8] = *(const h16*)(base + pb2 + (int)o1[i]);
  }
  return f;
}

// ---------------------------------------------------------------------------
// K0: weight prep (f32 -> padded f16 matrices in workspace)
// ---------------------------------------------------------------------------
__global__ void __launch_bounds__(256)
prep_kernel(const float* c1w, const float* c2w, const float* fcw,
            const float* a1w1, h16* W1, h16* W2, h16* FCT, h16* A1T) {
  int i = blockIdx.x * 256 + threadIdx.x;
  if (i < 32 * 96) {                       // conv1 im2col weights [32][96]
    int o = i / 96, k = i - o * 96;
    W1[i] = (h16)((o < 20 && k < 75) ? c1w[o * 75 + k] : 0.0f);
    return;
  }
  i -= 32 * 96;
  if (i < 64 * 512) {                      // conv2 im2col weights [64][512]
    int o = i >> 9, k = i & 511;
    W2[i] = (h16)((o < 50 && k < 500) ? c2w[o * 500 + k] : 0.0f);
    return;
  }
  i -= 64 * 512;
  if (i < 512 * 800) {                     // fc_w^T: [512 l][800 k]
    int l = i / 800, k = i - l * 800;
    FCT[i] = (h16)((l < 500) ? fcw[k * 500 + l] : 0.0f);
    return;
  }
  i -= 512 * 800;
  if (i < 128 * 512) {                     // a1_w1^T: [128 d][512 k]
    int d = i >> 9, k = i & 511;
    A1T[i] = (h16)((k < 500) ? a1w1[k * 128 + d] : 0.0f);
  }
}

// ---------------------------------------------------------------------------
// K1: conv1 + bias + relu + 2x2 maxpool.  One block per instance.
// GEMM [32(x20)][96] x [96][576]; image padded to 4 channels (ch3 = zeros) so
// padded K decodes land in zeros with no branch.  C rows >=20 go to a scratch
// row (21st) so stores are unconditional.
// LDS: img 6272 | offt 192 | W1 6144 | bias 128 | conv f32 [21][576] 48384
//      = 61120 B
// ---------------------------------------------------------------------------
__global__ void __launch_bounds__(256)
conv1_kernel(const float* x, const h16* W1, const float* b1, h16* H1) {
  extern __shared__ char smem[];
  h16*            img    = (h16*)smem;                     // 3136 halfs (4 ch)
  unsigned short* offt   = (unsigned short*)(smem + 6272); // 96 u16
  h16*            w1s    = (h16*)(smem + 6464);            // 3072 halfs
  float*          bias1s = (float*)(smem + 12608);         // 32
  float*          cb     = (float*)(smem + 12736);         // 21*576
  int n = blockIdx.x;
  int tid = threadIdx.x, lane = tid & 31, w = tid >> 5;
  const float* xn = x + (size_t)n * 2352;

  // image f32 -> f16 (vectorized), pad channel 3 with zeros
  for (int i = tid; i < 588; i += 256) {
    float4 v = ((const float4*)xn)[i];
    v4h h; h[0] = (h16)v.x; h[1] = (h16)v.y; h[2] = (h16)v.z; h[3] = (h16)v.w;
    *(v4h*)(img + i * 4) = h;
  }
  for (int i = 2352 + tid; i < 3136; i += 256) img[i] = (h16)0.0f;
  for (int i = tid; i < 384; i += 256)
    *(v8h*)(w1s + i * 8) = *(const v8h*)(W1 + i * 8);
  if (tid < 96) {  // byte offset table: k -> 2*(c*784 + ky*28 + kx)
    int k = tid, c = k / 25, rem = k - c * 25, ky = rem / 5, kx = rem - ky * 5;
    offt[k] = (unsigned short)((c * 784 + ky * 28 + kx) * 2);
  }
  if (tid < 32) bias1s[tid] = (tid < 20) ? b1[tid] : 0.0f;
  __syncthreads();

  int hb8 = (lane >> 4) << 3;
  // 72 tile jobs (2 M-tiles x 36 N-tiles), 9 per wave, uniform.
  for (int t = 0; t < 9; ++t) {
    int j = w + t * 8;
    int mt = j & 1, nt = j >> 1;
    int p = nt * 16 + (lane & 15);
    int oy = p / 24, ox = p - oy * 24;
    int pb2 = (oy * 28 + ox) * 2;
    v8f acc = {};
#pragma unroll
    for (int ks = 0; ks < 3; ++ks) {
      u16x8 o0 = *(const u16x8*)(offt + ks * 32 + hb8);
      u16x8 o1 = *(const u16x8*)(offt + ks * 32 + hb8 + 16);
      v16h af = ldfrag(w1s, 96, mt * 16, ks * 32, lane);
      v16h bf = gather16((const char*)img, pb2, o0, o1);
      acc = wmma16(af, bf, acc);
    }
    int col = nt * 16 + (lane & 15);
    int hi  = lane >> 4;
#pragma unroll
    for (int v = 0; v < 8; ++v) {
      int m  = mt * 16 + hi * 8 + v;
      int ms = (m < 20) ? m : 20;                 // scratch row, branch-free
      cb[ms * 576 + col] = acc[v] + bias1s[m];
    }
  }
  __syncthreads();

  for (int q = tid; q < 2880; q += 256) {      // pool 24x24 -> 12x12, relu
    int c = q / 144, r = q - c * 144, y = r / 12, xx = r - y * 12;
    const float* p = cb + c * 576 + (y * 2) * 24 + xx * 2;
    float m4 = fmaxf(fmaxf(p[0], p[1]), fmaxf(p[24], p[25]));
    H1[(size_t)n * 2880 + q] = (h16)fmaxf(m4, 0.0f);
  }
}

// ---------------------------------------------------------------------------
// K2: conv2 + bias + relu + maxpool -> flat X[N][800].  One block per instance.
// GEMM [64(x50)][512] x [512][64]; h1 padded to 21 channels (ch20 = zeros).
// Weights staged in two 32KB halves.  C rows >=50 go to scratch row 50.
// LDS: h1 6048 | offt2 1024 | w2 32768 | bias 256 | conv f32 [51][64] 13056
//      = 53152 B
// ---------------------------------------------------------------------------
__global__ void __launch_bounds__(256)
conv2_kernel(const h16* H1, const h16* W2, const float* b2, h16* Xout) {
  extern __shared__ char smem[];
  h16*            h1s    = (h16*)smem;                     // 3024 halfs (21 ch)
  unsigned short* offt2  = (unsigned short*)(smem + 6048); // 512 u16
  h16*            w2s    = (h16*)(smem + 7072);            // 64*256 halfs
  float*          bias2s = (float*)(smem + 39840);         // 64
  float*          cb     = (float*)(smem + 40096);         // 51*64
  int n = blockIdx.x;
  int tid = threadIdx.x, lane = tid & 31, w = tid >> 5;

  for (int i = tid; i < 360; i += 256)
    *(v8h*)(h1s + i * 8) = *(const v8h*)(H1 + (size_t)n * 2880 + i * 8);
  for (int i = 2880 + tid; i < 3024; i += 256) h1s[i] = (h16)0.0f;
  for (int k = tid; k < 512; k += 256) {  // k -> 2*(c*144 + ky*12 + kx)
    int c = k / 25, rem = k - c * 25, ky = rem / 5, kx = rem - ky * 5;
    offt2[k] = (unsigned short)((c * 144 + ky * 12 + kx) * 2);
  }
  if (tid < 64) bias2s[tid] = (tid < 50) ? b2[tid] : 0.0f;

  int mt  = w >> 1;
  int nt0 = (2 * w) & 3, nt1 = (2 * w + 1) & 3;
  int p0 = nt0 * 16 + (lane & 15), p1 = nt1 * 16 + (lane & 15);
  int pb20 = ((p0 >> 3) * 12 + (p0 & 7)) * 2;
  int pb21 = ((p1 >> 3) * 12 + (p1 & 7)) * 2;
  int hb8 = (lane >> 4) << 3;
  v8f acc0 = {}, acc1 = {};

  for (int hh = 0; hh < 2; ++hh) {
    __syncthreads();
#pragma unroll
    for (int t = 0; t < 8; ++t) {            // stage 16384 halfs (vectorized)
      int idx = (tid + t * 256) * 8;
      int r = idx >> 8, c = idx & 255;
      *(v8h*)(w2s + idx) = *(const v8h*)(W2 + r * 512 + hh * 256 + c);
    }
    __syncthreads();
#pragma unroll
    for (int ks = 0; ks < 8; ++ks) {
      int k0g = hh * 256 + ks * 32;
      u16x8 o0 = *(const u16x8*)(offt2 + k0g + hb8);
      u16x8 o1 = *(const u16x8*)(offt2 + k0g + hb8 + 16);
      v16h af  = ldfrag(w2s, 256, mt * 16, ks * 32, lane);
      v16h bf0 = gather16((const char*)h1s, pb20, o0, o1);
      v16h bf1 = gather16((const char*)h1s, pb21, o0, o1);
      acc0 = wmma16(af, bf0, acc0);
      acc1 = wmma16(af, bf1, acc1);
    }
  }

  int hi = lane >> 4;
#pragma unroll
  for (int v = 0; v < 8; ++v) {
    int m  = mt * 16 + hi * 8 + v;
    int ms = (m < 50) ? m : 50;                   // scratch row, branch-free
    float b = bias2s[m];
    cb[ms * 64 + nt0 * 16 + (lane & 15)] = acc0[v] + b;
    cb[ms * 64 + nt1 * 16 + (lane & 15)] = acc1[v] + b;
  }
  __syncthreads();

  for (int q = tid; q < 800; q += 256) {     // pool 8x8 -> 4x4, relu, flatten
    int c = q >> 4, r = q & 15, y = r >> 2, xx = r & 3;
    const float* p = cb + c * 64 + (y * 2) * 8 + xx * 2;
    float m4 = fmaxf(fmaxf(p[0], p[1]), fmaxf(p[8], p[9]));
    Xout[(size_t)n * 800 + q] = (h16)fmaxf(m4, 0.0f);
  }
}

// ---------------------------------------------------------------------------
// K3: H = relu(X[4096][800] @ fc_w[800][500] + b) -> f16 [4096][512].
// Grid (64,4): 64x128 output tile per block, 25 K-steps.  LDS 12288B.
// H pad columns (500..511) are don't-care downstream (A1T rows zeroed,
// segpool reads l<500), so the store is unconditional with a clamped bias.
// ---------------------------------------------------------------------------
__global__ void __launch_bounds__(256)
fc_kernel(const h16* X, const h16* FCT, const float* fcb, h16* H) {
  extern __shared__ char smem[];
  h16* As = (h16*)smem;             // 64*32
  h16* Bs = (h16*)(smem + 4096);    // 128*32
  int tid = threadIdx.x, lane = tid & 31, w = tid >> 5;
  int m0 = blockIdx.x * 64, n0 = blockIdx.y * 128;
  int mt = w >> 1;
  v8f acc[4] = {};

  for (int ks = 0; ks < 25; ++ks) {
    __syncthreads();
    {
      int idx = tid * 8;            // stage A: 64x32
      int r = idx >> 5, c = idx & 31;
      *(v8h*)(As + idx) =
          *(const v8h*)(X + (size_t)(m0 + r) * 800 + ks * 32 + c);
    }
#pragma unroll
    for (int t = 0; t < 2; ++t) {   // stage B: 128x32
      int idx = tid * 16 + t * 8;
      int r = idx >> 5, c = idx & 31;
      *(v8h*)(Bs + idx) =
          *(const v8h*)(FCT + (size_t)(n0 + r) * 800 + ks * 32 + c);
    }
    if (ks < 24)
      __builtin_prefetch(X + (size_t)(m0 + (tid >> 2)) * 800 + (ks + 1) * 32,
                         0, 1);
    __syncthreads();
    v16h af = ldfrag(As, 32, mt * 16, 0, lane);
#pragma unroll
    for (int jj = 0; jj < 4; ++jj) {
      int nt = (w & 1) * 4 + jj;
      v16h bf = ldfrag(Bs, 32, nt * 16, 0, lane);
      acc[jj] = wmma16(af, bf, acc[jj]);
    }
  }

  int hi = lane >> 4;
#pragma unroll
  for (int jj = 0; jj < 4; ++jj) {
    int nt  = (w & 1) * 4 + jj;
    int col = n0 + nt * 16 + (lane & 15);
    float b = fcb[(col < 500) ? col : 499];   // clamped, unconditional
#pragma unroll
    for (int v = 0; v < 8; ++v) {
      int row = m0 + mt * 16 + hi * 8 + v;
      H[(size_t)row * 512 + col] = (h16)fmaxf(acc[jj][v] + b, 0.0f);
    }
  }
}

// ---------------------------------------------------------------------------
// K4: a = tanh(H @ a1_w1 + b1) @ a1_w2 + b2  (per-instance scalar scores).
// Grid 64: 64 rows x 128 cols per block, 16 K-steps.  LDS 12544B.
// ---------------------------------------------------------------------------
__global__ void __launch_bounds__(256)
attn1_kernel(const h16* H, const h16* A1T, const float* b1, const float* w2,
             const float* b2, float* aOut, float* dA) {
  extern __shared__ char smem[];
  h16*   As   = (h16*)smem;             // 64*32
  h16*   Bs   = (h16*)(smem + 4096);    // 128*32
  float* aAcc = (float*)(smem + 12288); // 64
  int tid = threadIdx.x, lane = tid & 31, w = tid >> 5;
  int m0 = blockIdx.x * 64;
  if (tid < 64) aAcc[tid] = 0.0f;
  int mt = w >> 1;
  v8f acc[4] = {};

  for (int ks = 0; ks < 16; ++ks) {
    __syncthreads();
    {
      int idx = tid * 8;
      int r = idx >> 5, c = idx & 31;
      *(v8h*)(As + idx) =
          *(const v8h*)(H + (size_t)(m0 + r) * 512 + ks * 32 + c);
    }
#pragma unroll
    for (int t = 0; t < 2; ++t) {
      int idx = tid * 16 + t * 8;
      int r = idx >> 5, c = idx & 31;
      *(v8h*)(Bs + idx) =
          *(const v8h*)(A1T + (size_t)r * 512 + ks * 32 + c);
    }
    __syncthreads();
    v16h af = ldfrag(As, 32, mt * 16, 0, lane);
#pragma unroll
    for (int jj = 0; jj < 4; ++jj) {
      int nt = (w & 1) * 4 + jj;
      v16h bf = ldfrag(Bs, 32, nt * 16, 0, lane);
      acc[jj] = wmma16(af, bf, acc[jj]);
    }
  }

  int hi = lane >> 4;
#pragma unroll
  for (int jj = 0; jj < 4; ++jj) {
    int nt  = (w & 1) * 4 + jj;
    int col = nt * 16 + (lane & 15);        // 0..127
    float wv = w2[col];
    float s[8];
#pragma unroll
    for (int v = 0; v < 8; ++v) s[v] = tanhf(acc[jj][v] + b1[col]) * wv;
#pragma unroll
    for (int msk = 1; msk <= 8; msk <<= 1)
#pragma unroll
      for (int v = 0; v < 8; ++v) s[v] += __shfl_xor(s[v], msk, 32);
    if ((lane & 15) == 0)
#pragma unroll
      for (int v = 0; v < 8; ++v)
        atomicAdd(&aAcc[mt * 16 + hi * 8 + v], s[v]);
  }
  __syncthreads();
  if (tid < 64) {
    float a = aAcc[tid] + b2[0];
    aOut[m0 + tid] = a;
    dA[m0 + tid]   = a;   // third output: raw attn scores A (1,N)
  }
}

// ---------------------------------------------------------------------------
// K5: ragged per-segment softmax + weighted sum of H -> instance_f[64][500]
// ---------------------------------------------------------------------------
__global__ void __launch_bounds__(256)
segpool_kernel(const float* a, const int* idx, const h16* H, float* instF) {
  extern __shared__ char smem[];
  float* red = (float*)smem;     // 256
  int s = blockIdx.x, tid = threadIdx.x;
  int start = idx[s], end = idx[s + 1];

  float mx = -3.4e38f;
  for (int i = start + tid; i < end; i += 256) mx = fmaxf(mx, a[i]);
  red[tid] = mx;
  __syncthreads();
  for (int off = 128; off; off >>= 1) {
    if (tid < off) red[tid] = fmaxf(red[tid], red[tid + off]);
    __syncthreads();
  }
  float amax = red[0];
  __syncthreads();

  float se = 0.0f;
  for (int i = start + tid; i < end; i += 256) se += expf(a[i] - amax);
  red[tid] = se;
  __syncthreads();
  for (int off = 128; off; off >>= 1) {
    if (tid < off) red[tid] += red[tid + off];
    __syncthreads();
  }
  float inv = 1.0f / red[0];

  for (int l = tid; l < 500; l += 256) {
    float acc = 0.0f;
    for (int i = start; i < end; ++i)
      acc += expf(a[i] - amax) * inv * (float)H[(size_t)i * 512 + l];
    instF[s * 500 + l] = acc;
  }
}

// ---------------------------------------------------------------------------
// K6: attn2 + softmax + bag feature + classifier.  Single block, f32.
// LDS: th 32768 | batt 256 | bw 256 | bag 2048 | red 1024 = 36352B
// ---------------------------------------------------------------------------
__global__ void __launch_bounds__(256)
bag_kernel(const float* instF, const float* a2w1, const float* a2b1,
           const float* a2w2, const float* a2b2, const float* clfw,
           const float* clfb, float* out01) {
  extern __shared__ char smem[];
  float* th   = (float*)smem;       // 64*128
  float* batt = th + 8192;          // 64
  float* bw   = batt + 64;          // 64
  float* bag  = bw + 64;            // 512 (500 used)
  float* red  = bag + 512;          // 256
  int tid = threadIdx.x;

  for (int i = tid; i < 64 * 128; i += 256) {
    int s = i >> 7, d = i & 127;
    float acc = a2b1[d];
    for (int k = 0; k < 500; ++k) acc += instF[s * 500 + k] * a2w1[k * 128 + d];
    th[i] = tanhf(acc);
  }
  __syncthreads();
  if (tid < 64) {
    float acc = a2b2[0];
    for (int d = 0; d < 128; ++d) acc += th[tid * 128 + d] * a2w2[d];
    batt[tid] = acc;
  }
  __syncthreads();
  if (tid == 0) {
    float mx = -3.4e38f;
    for (int s = 0; s < 64; ++s) mx = fmaxf(mx, batt[s]);
    float sum = 0.0f;
    for (int s = 0; s < 64; ++s) { float e = expf(batt[s] - mx); bw[s] = e; sum += e; }
    float inv = 1.0f / sum;
    for (int s = 0; s < 64; ++s) bw[s] *= inv;
  }
  __syncthreads();
  for (int l = tid; l < 500; l += 256) {
    float acc = 0.0f;
    for (int s = 0; s < 64; ++s) acc += bw[s] * instF[s * 500 + l];
    bag[l] = acc;
  }
  __syncthreads();
  float part = 0.0f;
  for (int l = tid; l < 500; l += 256) part += bag[l] * clfw[l];
  red[tid] = part;
  __syncthreads();
  for (int off = 128; off; off >>= 1) {
    if (tid < off) red[tid] += red[tid + off];
    __syncthreads();
  }
  if (tid == 0) {
    float logit = red[0] + clfb[0];
    float p = 1.0f / (1.0f + expf(-logit));
    out01[0] = p;
    out01[1] = (p >= 0.5f) ? 1.0f : 0.0f;
  }
}

// ---------------------------------------------------------------------------
extern "C" void kernel_launch(void* const* d_in, const int* in_sizes, int n_in,
                              void* d_out, int out_size, void* d_ws,
                              size_t ws_size, hipStream_t stream) {
  (void)in_sizes; (void)n_in; (void)out_size; (void)ws_size;
  const float* x    = (const float*)d_in[0];
  const int*   idx  = (const int*)d_in[1];
  const float* c1w  = (const float*)d_in[2];
  const float* c1b  = (const float*)d_in[3];
  const float* c2w  = (const float*)d_in[4];
  const float* c2b  = (const float*)d_in[5];
  const float* fcw  = (const float*)d_in[6];
  const float* fcb  = (const float*)d_in[7];
  const float* a1w1 = (const float*)d_in[8];
  const float* a1b1 = (const float*)d_in[9];
  const float* a1w2 = (const float*)d_in[10];
  const float* a1b2 = (const float*)d_in[11];
  const float* a2w1 = (const float*)d_in[12];
  const float* a2b1 = (const float*)d_in[13];
  const float* a2w2 = (const float*)d_in[14];
  const float* a2b2 = (const float*)d_in[15];
  const float* clfw = (const float*)d_in[16];
  const float* clfb = (const float*)d_in[17];
  float* out = (float*)d_out;
  char*  ws  = (char*)d_ws;

  // workspace layout (~30 MB; big conv1 buffer aliased by later buffers)
  h16* W1  = (h16*)(ws);                          //   6144 B
  h16* W2  = (h16*)(ws + 8192);                   //  65536 B
  h16* FCT = (h16*)(ws + 8192 + 65536);           // 819200 B
  h16* A1T = (h16*)(ws + 8192 + 65536 + 819200);  // 131072 B  (ends < 1MiB)
  h16* X   = (h16*)(ws + (1u << 20));             // 6553600 B
  char* big = ws + (1u << 20) + 6553600;          // 23592960 B region
  h16*   H1    = (h16*)big;                       // live K1->K2
  h16*   H     = (h16*)big;                       // live K3->K5 (aliases H1)
  float* aS    = (float*)(big + 4194304);         // 16384 B
  float* instF = (float*)(big + 4194304 + 16384); // 128000 B

  prep_kernel<<<1996, 256, 0, stream>>>(c1w, c2w, fcw, a1w1, W1, W2, FCT, A1T);
  conv1_kernel<<<4096, 256, 61120, stream>>>(x, W1, c1b, H1);
  conv2_kernel<<<4096, 256, 53152, stream>>>(H1, W2, c2b, X);
  fc_kernel<<<dim3(64, 4), 256, 12288, stream>>>(X, FCT, fcb, H);
  attn1_kernel<<<64, 256, 12544, stream>>>(H, A1T, a1b1, a1w2, a1b2, aS,
                                           out + 2);
  segpool_kernel<<<64, 256, 1024, stream>>>(aS, idx, H, instF);
  bag_kernel<<<1, 256, 36352, stream>>>(instF, a2w1, a2b1, a2w2, a2b2, clfw,
                                        clfb, out);
}
